// RnnModelInterp_2181843386984
// MI455X (gfx1250) — matrix operand order repaired and checked by reading, not compiled, not run on hardware
//
#include <hip/hip_runtime.h>
#include <hip/hip_bf16.h>

// ---------------- problem constants ----------------
static constexpr int kT = 128;   // timesteps
static constexpr int kB = 8192;  // batch
static constexpr int kM = 64;    // measures
static constexpr int kH = 256;   // hidden
static constexpr int kG = 4 * kH; // 1024 gate columns
static constexpr int MT = 32;    // batch rows per workgroup (2 WMMA M-tiles)

// LDS strides (elements), padded; bf16 row strides are multiples of 8 elems
// so v8bf (16B) loads stay 16B-aligned.
static constexpr int GS = kG + 4;   // gates fp32 stride (1028)
static constexpr int CS = kH + 4;   // cell fp32 stride (260)
static constexpr int HS = kH + 8;   // h bf16 stride (264)
static constexpr int XS = kM + 8;   // x bf16 stride (72)
static constexpr int OS = kM + 4;   // o fp32 stride (68)

typedef __attribute__((ext_vector_type(8)))  __bf16 v8bf;
typedef __attribute__((ext_vector_type(16))) __bf16 v16bf;
typedef __attribute__((ext_vector_type(8)))  float  v8f;

__device__ __forceinline__ __bf16 tobf(float f) {
  union { float f; unsigned u; } a; a.f = f;
  unsigned u = a.u + 0x7FFFu + ((a.u >> 16) & 1u);   // round-to-nearest-even
  unsigned short s = (unsigned short)(u >> 16);
  __bf16 r; __builtin_memcpy(&r, &s, 2);
  return r;
}

__device__ __forceinline__ float sigf(float x) { return 1.0f / (1.0f + __expf(-x)); }

__device__ __forceinline__ v16bf cat16(v8bf a, v8bf b) {
  return __builtin_shufflevector(a, b, 0,1,2,3,4,5,6,7,8,9,10,11,12,13,14,15);
}

// A fragment (16x32 bf16): contiguous k order, same as load_b_frag.
// row = &lds[m*stride]; lanes 0-15 take K {kt*32+0..15},
// lanes 16-31 take K {kt*32+16..31}.  -> two ds_load_b128
__device__ __forceinline__ v16bf load_a_frag(const __bf16* row, int kt, int hi) {
  const __bf16* p = row + kt * 32 + hi * 16;
  v8bf lo = *(const v8bf*)p;
  v8bf hh = *(const v8bf*)(p + 8);
  return cat16(lo, hh);
}

// B fragment (32x16 bf16) from row-major W[4H][K] (so B = W^T slice):
// lane n = ncol holds K = kt*32 + hi*16 .. +15 (16 contiguous bf16)
// -> two global_load_b128 per lane.
__device__ __forceinline__ v16bf load_b_frag(const __bf16* __restrict__ W, int K,
                                             int ncol, int kt, int hi) {
  const __bf16* p = W + (size_t)ncol * K + kt * 32 + hi * 16;
  v8bf lo = *(const v8bf*)p;
  v8bf hh = *(const v8bf*)(p + 8);
  return cat16(lo, hh);
}

__device__ __forceinline__ v8f wmma_bf16(v16bf a, v16bf b, v8f c) {
  return __builtin_amdgcn_wmma_f32_16x16x32_bf16(false, a, false, b, (short)0, c,
                                                 false, false);
}

// -------- weight prep: fp32 -> bf16, bias pair pre-sum --------
__global__ __launch_bounds__(256)
void prep_kernel(const float* __restrict__ Wih0, const float* __restrict__ Whh0,
                 const float* __restrict__ Wih1, const float* __restrict__ Whh1,
                 const float* __restrict__ Wout,
                 const float* __restrict__ bih0, const float* __restrict__ bhh0,
                 const float* __restrict__ bih1, const float* __restrict__ bhh1,
                 __bf16* __restrict__ wih0b, __bf16* __restrict__ whh0b,
                 __bf16* __restrict__ wih1b, __bf16* __restrict__ whh1b,
                 __bf16* __restrict__ woutb,
                 float* __restrict__ b0, float* __restrict__ b1) {
  int i = blockIdx.x * blockDim.x + threadIdx.x;
  if (i < kG * kM)  wih0b[i] = tobf(Wih0[i]);
  if (i < kG * kH) {
    whh0b[i] = tobf(Whh0[i]);
    wih1b[i] = tobf(Wih1[i]);
    whh1b[i] = tobf(Whh1[i]);
  }
  if (i < kM * kH)  woutb[i] = tobf(Wout[i]);
  if (i < kG) { b0[i] = bih0[i] + bhh0[i]; b1[i] = bih1[i] + bhh1[i]; }
}

// -------- main persistent-over-time LSTM kernel --------
__global__ __launch_bounds__(256)
void lstm_kernel(const float* __restrict__ val,
                 const __bf16* __restrict__ Wih0, const __bf16* __restrict__ Whh0,
                 const __bf16* __restrict__ Wih1, const __bf16* __restrict__ Whh1,
                 const __bf16* __restrict__ Wout,
                 const float* __restrict__ b0sum, const float* __restrict__ b1sum,
                 const float* __restrict__ bout,
                 float* __restrict__ out) {
  extern __shared__ char smem[];
  float*  g   = (float*)smem;              // MT*GS fp32 gate buffer
  float*  c1  = g  + MT * GS;              // MT*CS
  float*  c2  = c1 + MT * CS;              // MT*CS
  float*  ob  = c2 + MT * CS;              // MT*OS
  __bf16* h1  = (__bf16*)(ob + MT * OS);   // MT*HS
  __bf16* h2  = h1 + MT * HS;              // MT*HS
  __bf16* xb  = h2 + MT * HS;              // MT*XS

  const int tid  = threadIdx.x;
  const int lane = tid & 31;
  const int wid  = tid >> 5;
  const int mrow = lane & 15;
  const int hi   = lane >> 4;
  const int mb   = hi * 8;                 // C/D row group base for this lane
  const int r0   = blockIdx.x * MT;        // batch-tile base row

  // zero-init recurrent state
  const __bf16 zb = tobf(0.0f);
  for (int i = tid; i < MT * HS; i += 256) { h1[i] = zb; h2[i] = zb; }
  for (int i = tid; i < MT * CS; i += 256) { c1[i] = 0.0f; c2[i] = 0.0f; }

  // x0 = val[0] with NaN -> 0
  for (int e = tid; e < MT * kM; e += 256) {
    int r = e >> 6, mc = e & 63;
    float v = val[(size_t)(r0 + r) * kM + mc];
    if (v != v) v = 0.0f;
    xb[r * XS + mc] = tobf(v);
  }
  __syncthreads();

  for (int t = 1; t < kT; ++t) {
    // ============ layer 0: gates0 = x@Wih0^T + h1@Whh0^T + b ============
    // pass A: seed bias, add x@Wih0^T (K=64). B frags in regs, reused over mt.
    #pragma clang loop unroll(disable)
    for (int nt = 0; nt < 8; ++nt) {
      const int ncol = (wid * 8 + nt) * 16 + mrow;
      v16bf bfr[2];
      #pragma unroll
      for (int kt = 0; kt < 2; ++kt) bfr[kt] = load_b_frag(Wih0, kM, ncol, kt, hi);
      const float bias = b0sum[ncol];
      #pragma unroll
      for (int mt = 0; mt < 2; ++mt) {
        const __bf16* ar = xb + (mt * 16 + mrow) * XS;
        v8f acc;
        #pragma unroll
        for (int i2 = 0; i2 < 8; ++i2) acc[i2] = bias;
        #pragma unroll
        for (int kt = 0; kt < 2; ++kt)
          acc = wmma_bf16(load_a_frag(ar, kt, hi), bfr[kt], acc);
        #pragma unroll
        for (int i2 = 0; i2 < 8; ++i2) g[(mt * 16 + mb + i2) * GS + ncol] = acc[i2];
      }
    }
    // pass B: += h1@Whh0^T (K=256). Same-wave LDS ops are in-order and each
    // wave reads back only its own partials -> no barrier needed between passes.
    #pragma clang loop unroll(disable)
    for (int nt = 0; nt < 8; ++nt) {
      const int ncol = (wid * 8 + nt) * 16 + mrow;
      v16bf bfr[8];
      #pragma unroll
      for (int kt = 0; kt < 8; ++kt) bfr[kt] = load_b_frag(Whh0, kH, ncol, kt, hi);
      #pragma unroll
      for (int mt = 0; mt < 2; ++mt) {
        const __bf16* ar = h1 + (mt * 16 + mrow) * HS;
        v8f acc;
        #pragma unroll
        for (int i2 = 0; i2 < 8; ++i2) acc[i2] = g[(mt * 16 + mb + i2) * GS + ncol];
        #pragma unroll
        for (int kt = 0; kt < 8; ++kt)
          acc = wmma_bf16(load_a_frag(ar, kt, hi), bfr[kt], acc);
        #pragma unroll
        for (int i2 = 0; i2 < 8; ++i2) g[(mt * 16 + mb + i2) * GS + ncol] = acc[i2];
      }
    }
    __syncthreads();

    // activations -> c1, h1   (thread = hidden unit column)
    {
      const int j = tid;
      #pragma unroll 4
      for (int r = 0; r < MT; ++r) {
        float gi = g[r * GS + j];
        float gf = g[r * GS + kH + j];
        float gg = g[r * GS + 2 * kH + j];
        float go = g[r * GS + 3 * kH + j];
        float c  = sigf(gf) * c1[r * CS + j] + sigf(gi) * tanhf(gg);
        c1[r * CS + j] = c;
        h1[r * HS + j] = tobf(sigf(go) * tanhf(c));
      }
    }
    __syncthreads();

    // ============ layer 1: gates1 = h1@Wih1^T + h2@Whh1^T + b ============
    // pass A: seed bias, add h1@Wih1^T
    #pragma clang loop unroll(disable)
    for (int nt = 0; nt < 8; ++nt) {
      const int ncol = (wid * 8 + nt) * 16 + mrow;
      v16bf bfr[8];
      #pragma unroll
      for (int kt = 0; kt < 8; ++kt) bfr[kt] = load_b_frag(Wih1, kH, ncol, kt, hi);
      const float bias = b1sum[ncol];
      #pragma unroll
      for (int mt = 0; mt < 2; ++mt) {
        const __bf16* ar = h1 + (mt * 16 + mrow) * HS;
        v8f acc;
        #pragma unroll
        for (int i2 = 0; i2 < 8; ++i2) acc[i2] = bias;
        #pragma unroll
        for (int kt = 0; kt < 8; ++kt)
          acc = wmma_bf16(load_a_frag(ar, kt, hi), bfr[kt], acc);
        #pragma unroll
        for (int i2 = 0; i2 < 8; ++i2) g[(mt * 16 + mb + i2) * GS + ncol] = acc[i2];
      }
    }
    // pass B: += h2@Whh1^T
    #pragma clang loop unroll(disable)
    for (int nt = 0; nt < 8; ++nt) {
      const int ncol = (wid * 8 + nt) * 16 + mrow;
      v16bf bfr[8];
      #pragma unroll
      for (int kt = 0; kt < 8; ++kt) bfr[kt] = load_b_frag(Whh1, kH, ncol, kt, hi);
      #pragma unroll
      for (int mt = 0; mt < 2; ++mt) {
        const __bf16* ar = h2 + (mt * 16 + mrow) * HS;
        v8f acc;
        #pragma unroll
        for (int i2 = 0; i2 < 8; ++i2) acc[i2] = g[(mt * 16 + mb + i2) * GS + ncol];
        #pragma unroll
        for (int kt = 0; kt < 8; ++kt)
          acc = wmma_bf16(load_a_frag(ar, kt, hi), bfr[kt], acc);
        #pragma unroll
        for (int i2 = 0; i2 < 8; ++i2) g[(mt * 16 + mb + i2) * GS + ncol] = acc[i2];
      }
    }
    __syncthreads();

    // activations -> c2, h2
    {
      const int j = tid;
      #pragma unroll 4
      for (int r = 0; r < MT; ++r) {
        float gi = g[r * GS + j];
        float gf = g[r * GS + kH + j];
        float gg = g[r * GS + 2 * kH + j];
        float go = g[r * GS + 3 * kH + j];
        float c  = sigf(gf) * c2[r * CS + j] + sigf(gi) * tanhf(gg);
        c2[r * CS + j] = c;
        h2[r * HS + j] = tobf(sigf(go) * tanhf(c));
      }
    }
    __syncthreads();

    // ============ output projection: o = h2@Wout^T + bout ============
    // 8 tile-jobs (4 N-tiles x 2 M-tiles) -> one per wave, all waves busy.
    {
      const int ntO = wid & 3, mtO = wid >> 2;
      const int ncol = ntO * 16 + mrow;
      v16bf bfr[8];
      #pragma unroll
      for (int kt = 0; kt < 8; ++kt) bfr[kt] = load_b_frag(Wout, kH, ncol, kt, hi);
      const __bf16* ar = h2 + (mtO * 16 + mrow) * HS;
      const float bias = bout[ncol];
      v8f acc;
      #pragma unroll
      for (int i2 = 0; i2 < 8; ++i2) acc[i2] = bias;
      #pragma unroll
      for (int kt = 0; kt < 8; ++kt)
        acc = wmma_bf16(load_a_frag(ar, kt, hi), bfr[kt], acc);
      #pragma unroll
      for (int i2 = 0; i2 < 8; ++i2) ob[(mtO * 16 + mb + i2) * OS + ncol] = acc[i2];
    }
    __syncthreads();

    // =========== emit out[t-1], impute next x from val[t] ==========
    #pragma unroll
    for (int e = tid; e < MT * kM; e += 256) {
      int r = e >> 6, mc = e & 63;
      float ov = ob[r * OS + mc];
      out[((size_t)(t - 1) * kB + (r0 + r)) * kM + mc] = ov;
      float v = val[((size_t)t * kB + (r0 + r)) * kM + mc];
      xb[r * XS + mc] = tobf((v != v) ? ov : v);
    }
    __syncthreads();
  }
}

extern "C" void kernel_launch(void* const* d_in, const int* in_sizes, int n_in,
                              void* d_out, int out_size, void* d_ws, size_t ws_size,
                              hipStream_t stream) {
  (void)in_sizes; (void)n_in; (void)out_size; (void)ws_size;
  const float* val  = (const float*)d_in[0];
  // d_in[1] = inaux (unused by forward)
  const float* Wih0 = (const float*)d_in[2];
  const float* Whh0 = (const float*)d_in[3];
  const float* bih0 = (const float*)d_in[4];
  const float* bhh0 = (const float*)d_in[5];
  const float* Wih1 = (const float*)d_in[6];
  const float* Whh1 = (const float*)d_in[7];
  const float* bih1 = (const float*)d_in[8];
  const float* bhh1 = (const float*)d_in[9];
  const float* Wout = (const float*)d_in[10];
  const float* bout = (const float*)d_in[11];
  float* out = (float*)d_out;

  // workspace layout: bf16 weights then fp32 bias sums
  __bf16* wih0b = (__bf16*)d_ws;                 // 1024*64
  __bf16* whh0b = wih0b + kG * kM;               // 1024*256
  __bf16* wih1b = whh0b + kG * kH;               // 1024*256
  __bf16* whh1b = wih1b + kG * kH;               // 1024*256
  __bf16* woutb = whh1b + kG * kH;               // 64*256
  float*  b0    = (float*)(woutb + kM * kH);
  float*  b1    = b0 + kG;

  prep_kernel<<<(kG * kH + 255) / 256, 256, 0, stream>>>(
      Wih0, Whh0, Wih1, Whh1, Wout, bih0, bhh0, bih1, bhh1,
      wih0b, whh0b, wih1b, whh1b, woutb, b0, b1);

  const size_t smem = (size_t)(MT * GS + 2 * MT * CS + MT * OS) * sizeof(float) +
                      (size_t)(2 * MT * HS + MT * XS) * sizeof(__bf16);

  lstm_kernel<<<kB / MT, 256, smem, stream>>>(
      val, wih0b, whh0b, wih1b, whh1b, woutb, b0, b1, bout, out);
}
